// GraphNet_47682726920887
// MI455X (gfx1250) — matrix operand order repaired and checked by reading, not compiled, not run on hardware
//
#include <hip/hip_runtime.h>
#include <hip/hip_bf16.h>

typedef __attribute__((ext_vector_type(16))) __bf16 v16bf;
typedef __attribute__((ext_vector_type(8)))  float  v8f;
typedef __attribute__((ext_vector_type(4)))  float  v4f;
typedef __attribute__((ext_vector_type(4)))  unsigned v4u;

#define HEADS 4

// ---------- helpers ----------
__device__ __forceinline__ unsigned short f32_to_bf16_rn(float f) {
    unsigned u = __float_as_uint(f);
    unsigned r = u + 0x7FFFu + ((u >> 16) & 1u);   // round-to-nearest-even
    return (unsigned short)(r >> 16);
}
__device__ __forceinline__ float bf16_to_f32(unsigned short h) {
    return __uint_as_float(((unsigned)h) << 16);
}
// monotone float <-> uint encoding for atomicMax over signed floats
__device__ __forceinline__ unsigned enc_f32(float f) {
    unsigned u = __float_as_uint(f);
    return (u & 0x80000000u) ? ~u : (u | 0x80000000u);
}
__device__ __forceinline__ float dec_f32(unsigned e) {
    unsigned u = (e & 0x80000000u) ? (e ^ 0x80000000u) : ~e;
    return __uint_as_float(u);
}
#define ENC_NEG_INF 0x007fffffu   // enc(-inf)

union BF16x16 { v16bf v; unsigned short s[16]; };

// ---------- generic fill ----------
__global__ void k_fill_u32(unsigned* __restrict__ p, unsigned v, int n) {
    int i = blockIdx.x * blockDim.x + threadIdx.x;
    if (i < n) p[i] = v;
}

// ---------- pack weight matrix (K x N fp32, row-major) into WMMA B-operand swizzle,
// ---------- split into bf16 hi/lo. Layout: ((kt*NTpad+nt)*32 + lane)*16 + j,
// ---------- lane<16: n=nt*16+lane, K=kt*32 + j ; lane>=16: n=nt*16+lane-16, K=kt*32+16+j
__global__ void k_pack_w(const float* __restrict__ W, int K, int N, int NTpad,
                         unsigned short* __restrict__ Whi, unsigned short* __restrict__ Wlo)
{
    int gid = blockIdx.x * blockDim.x + threadIdx.x;
    int KT = K >> 5;
    int total = KT * NTpad * 32;
    if (gid >= total) return;
    int lane = gid & 31;
    int nt   = (gid >> 5) % NTpad;
    int kt   = (gid >> 5) / NTpad;
    int n    = nt * 16 + (lane & 15);
    int kb   = (lane >> 4) * 16;
    size_t base = (size_t)gid * 16;
    for (int j = 0; j < 16; ++j) {
        int k = kt * 32 + kb + j;
        float v = (n < N && k < K) ? W[(size_t)k * N + n] : 0.0f;
        unsigned short h = f32_to_bf16_rn(v);
        Whi[base + j] = h;
        Wlo[base + j] = f32_to_bf16_rn(v - bf16_to_f32(h));
    }
}

// ---------- WMMA GEMM: C(MxN) = A(MxK fp32) * B(KxN, pre-packed bf16 hi/lo)
// one wave -> 16 rows x 64 cols (4 n-tiles), bf16x2 split => 3 wmma per tile per k-step
__global__ void k_gemm_wmma(const float* __restrict__ A,
                            const unsigned short* __restrict__ Bhi,
                            const unsigned short* __restrict__ Blo,
                            float* __restrict__ C,
                            int M, int K, int N, int NTpad)
{
    const int lane    = threadIdx.x & 31;
    const int gwave   = (blockIdx.x * blockDim.x + threadIdx.x) >> 5;
    const int ngroups = NTpad >> 2;
    const int mtiles  = M >> 4;
    if (gwave >= mtiles * ngroups) return;
    const int mt = gwave / ngroups;
    const int ng = gwave - mt * ngroups;
    const int m  = mt * 16 + (lane & 15);
    const int kb = (lane >> 4) << 4;          // A frag: lanes 16-31 carry K offset +16
    const int KT = K >> 5;

    v8f acc[4] = {};

    for (int kt = 0; kt < KT; ++kt) {
        // ---- build A fragment (bf16 hi/lo split) ----
        const float* arow = A + (size_t)m * K + kt * 32 + kb;
        v4f a0 = *(const v4f*)(arow);
        v4f a1 = *(const v4f*)(arow + 4);
        v4f a2 = *(const v4f*)(arow + 8);
        v4f a3 = *(const v4f*)(arow + 12);
        float av[16];
        #pragma unroll
        for (int i = 0; i < 4; ++i) { av[i]=a0[i]; av[4+i]=a1[i]; av[8+i]=a2[i]; av[12+i]=a3[i]; }
        BF16x16 ahi, alo;
        #pragma unroll
        for (int i = 0; i < 16; ++i) {
            unsigned short h = f32_to_bf16_rn(av[i]);
            ahi.s[i] = h;
            alo.s[i] = f32_to_bf16_rn(av[i] - bf16_to_f32(h));
        }
        // ---- 4 B tiles, 3 WMMAs each ----
        #pragma unroll
        for (int t = 0; t < 4; ++t) {
            size_t tb = ((size_t)(kt * NTpad + (ng * 4 + t))) * 512 + (size_t)lane * 16;
            BF16x16 bhi, blo;
            *(v4u*)&bhi.s[0] = *(const v4u*)(Bhi + tb);
            *(v4u*)&bhi.s[8] = *(const v4u*)(Bhi + tb + 8);
            *(v4u*)&blo.s[0] = *(const v4u*)(Blo + tb);
            *(v4u*)&blo.s[8] = *(const v4u*)(Blo + tb + 8);
            acc[t] = __builtin_amdgcn_wmma_f32_16x16x32_bf16(false, ahi.v, false, bhi.v,
                                                             (short)0, acc[t], false, false);
            acc[t] = __builtin_amdgcn_wmma_f32_16x16x32_bf16(false, alo.v, false, bhi.v,
                                                             (short)0, acc[t], false, false);
            acc[t] = __builtin_amdgcn_wmma_f32_16x16x32_bf16(false, ahi.v, false, blo.v,
                                                             (short)0, acc[t], false, false);
        }
    }
    // ---- store: lane -> col = (lane&15), rows mbase..mbase+7 ----
    const int mbase = mt * 16 + ((lane >> 4) << 3);
    #pragma unroll
    for (int t = 0; t < 4; ++t) {
        int n = (ng * 4 + t) * 16 + (lane & 15);
        if (n < N) {
            #pragma unroll
            for (int r = 0; r < 8; ++r)
                C[(size_t)(mbase + r) * N + n] = acc[t][r];
        }
    }
}

// ---------- per-node attention dot products ----------
__global__ void k_node_dots(const float* __restrict__ h, const float* __restrict__ att,
                            float* __restrict__ al, float* __restrict__ ar, int nNodes, int C)
{
    int gid = blockIdx.x * blockDim.x + threadIdx.x;
    if (gid >= nNodes * HEADS) return;
    int hd = gid & 3;
    int node = gid >> 2;
    const float* hp = h + (size_t)node * HEADS * C + (size_t)hd * C;
    const float* aL = att + (size_t)hd * 2 * C;
    const float* aR = aL + C;
    float sl = 0.f, sr = 0.f;
    for (int c = 0; c < C; ++c) { float v = hp[c]; sl += v * aL[c]; sr += v * aR[c]; }
    al[gid] = sl; ar[gid] = sr;
}

// ---------- edge pass 1: leaky-relu(alpha), segment max via encoded atomicMax ----------
__global__ void k_edge_alpha_max(const long long* __restrict__ ei, int nE, int nTot,
                                 const float* __restrict__ al, const float* __restrict__ ar,
                                 float* __restrict__ alphaBuf, unsigned* __restrict__ amax)
{
    int e = blockIdx.x * blockDim.x + threadIdx.x;
    if (e >= nTot) return;
    long long r, c;
    if (e < nE) { r = ei[e]; c = ei[nE + e]; } else { r = c = (long long)(e - nE); }
    #pragma unroll
    for (int hd = 0; hd < HEADS; ++hd) {
        float a = al[r * HEADS + hd] + ar[c * HEADS + hd];
        a = (a > 0.f) ? a : 0.2f * a;
        alphaBuf[(size_t)e * HEADS + hd] = a;
        atomicMax(&amax[r * HEADS + hd], enc_f32(a));
    }
}

// ---------- edge pass 2: exp(alpha - max), segment sum ----------
__global__ void k_edge_exp(const long long* __restrict__ ei, int nE, int nTot,
                           const unsigned* __restrict__ amax,
                           float* __restrict__ alphaBuf /* in alpha, out ex */,
                           float* __restrict__ denom)
{
    int e = blockIdx.x * blockDim.x + threadIdx.x;
    if (e >= nTot) return;
    long long r = (e < nE) ? ei[e] : (long long)(e - nE);
    #pragma unroll
    for (int hd = 0; hd < HEADS; ++hd) {
        float ex = expf(alphaBuf[(size_t)e * HEADS + hd] - dec_f32(amax[r * HEADS + hd]));
        alphaBuf[(size_t)e * HEADS + hd] = ex;
        atomicAdd(&denom[r * HEADS + hd], ex);
    }
}

// ---------- edge pass 3: scatter-add coef * feat[col] into out[row] ----------
__global__ void k_aggregate(const long long* __restrict__ ei, int nE, int nTot,
                            const float* __restrict__ exBuf, const float* __restrict__ denom,
                            const float* __restrict__ feat, float* __restrict__ out,
                            int C, int HC)
{
    int gid = blockIdx.x * blockDim.x + threadIdx.x;
    int qn = HC >> 2;                    // float4 groups per edge
    if (gid >= nTot * qn) return;
    int e  = gid / qn;
    int q  = gid - e * qn;
    int c0 = q * 4;
    int hd = c0 / C;
    long long r, c;
    if (e < nE) { r = ei[e]; c = ei[nE + e]; } else { r = c = (long long)(e - nE); }
    float coef = exBuf[(size_t)e * HEADS + hd] / (denom[r * HEADS + hd] + 1e-16f);
    v4f f = *(const v4f*)(feat + (size_t)c * HC + c0);
    float* o = out + (size_t)r * HC + c0;
    atomicAdd(o + 0, coef * f[0]);
    atomicAdd(o + 1, coef * f[1]);
    atomicAdd(o + 2, coef * f[2]);
    atomicAdd(o + 3, coef * f[3]);
}

// ---------- relu(x + bias) elementwise ----------
__global__ void k_relu_bias(float* __restrict__ x, const float* __restrict__ b, int total, int N)
{
    int i = blockIdx.x * blockDim.x + threadIdx.x;
    if (i < total) x[i] = fmaxf(x[i] + b[i % N], 0.f);
}

// ---------- mean over heads + bias -> final output ----------
__global__ void k_mean_bias(const float* __restrict__ agg, const float* __restrict__ b,
                            float* __restrict__ out, int nNodes, int C)
{
    int gid = blockIdx.x * blockDim.x + threadIdx.x;
    if (gid >= nNodes * C) return;
    int node = gid / C;
    int c = gid - node * C;
    const float* p = agg + (size_t)node * HEADS * C + c;
    out[gid] = 0.25f * (p[0] + p[C] + p[2 * C] + p[3 * C]) + b[c];
}

extern "C" void kernel_launch(void* const* d_in, const int* in_sizes, int n_in,
                              void* d_out, int out_size, void* d_ws, size_t ws_size,
                              hipStream_t stream)
{
    const float*     x    = (const float*)d_in[0];
    const long long* ei   = (const long long*)d_in[1];   // int64, [2,480000] flat
    const float*     w0   = (const float*)d_in[2];
    const float*     att0 = (const float*)d_in[3];
    const float*     b0   = (const float*)d_in[4];
    const float*     w1   = (const float*)d_in[5];
    const float*     att1 = (const float*)d_in[6];
    const float*     b1   = (const float*)d_in[7];
    float* outp = (float*)d_out;

    const int NN = 30000, NE = 480000, NT = 510000;
    const int HC0 = 256, HC1 = 160, C0 = 64, C1 = 40;

    // ---- workspace layout (buf0/buf1 reused across layers) ----
    char* ws = (char*)d_ws;
    float* buf0 = (float*)ws;                        // h0 (30000x256) then h1 (30000x160)
    float* buf1 = (float*)(ws + 30720000);           // out0 (30000x256) then agg1 (30000x160)
    float* al    = (float*)(ws + 61440000);          // 30000*4
    float* ar    = al + 120000;
    unsigned* amax = (unsigned*)(ar + 120000);
    float* denom = (float*)(amax + 120000);
    float* exbuf = denom + 120000;                   // 510000*4
    unsigned short* w0hi = (unsigned short*)(exbuf + 2040000);
    unsigned short* w0lo = w0hi + 65536;             // KT=8, NTpad=16 -> 8*16*512
    unsigned short* w1hi = w0lo + 65536;
    unsigned short* w1lo = w1hi + 49152;             // KT=8, NTpad=12 -> 8*12*512

    const int B = 256;
    #define GRID(n) (((n) + B - 1) / B)

    // ---- pack weights into WMMA B-operand bf16 hi/lo ----
    k_pack_w<<<GRID(8 * 16 * 32), B, 0, stream>>>(w0, 256, 256, 16, w0hi, w0lo);
    k_pack_w<<<GRID(8 * 12 * 32), B, 0, stream>>>(w1, 256, 160, 12, w1hi, w1lo);

    // ================= layer 0 =================
    int waves0 = (NN / 16) * (16 / 4);               // 7500 waves
    k_gemm_wmma<<<GRID(waves0 * 32), B, 0, stream>>>(x, w0hi, w0lo, buf0, NN, 256, 256, 16);

    k_node_dots<<<GRID(NN * HEADS), B, 0, stream>>>(buf0, att0, al, ar, NN, C0);

    k_fill_u32<<<GRID(NN * HC0), B, 0, stream>>>((unsigned*)buf1, 0u, NN * HC0);
    k_fill_u32<<<GRID(NN * HEADS), B, 0, stream>>>(amax, ENC_NEG_INF, NN * HEADS);
    k_fill_u32<<<GRID(NN * HEADS), B, 0, stream>>>((unsigned*)denom, 0u, NN * HEADS);

    k_edge_alpha_max<<<GRID(NT), B, 0, stream>>>(ei, NE, NT, al, ar, exbuf, amax);
    k_edge_exp<<<GRID(NT), B, 0, stream>>>(ei, NE, NT, amax, exbuf, denom);
    k_aggregate<<<GRID(NT * (HC0 / 4)), B, 0, stream>>>(ei, NE, NT, exbuf, denom,
                                                        buf0, buf1, C0, HC0);
    k_relu_bias<<<GRID(NN * HC0), B, 0, stream>>>(buf1, b0, NN * HC0, HC0);

    // ================= layer 1 =================
    int waves1 = (NN / 16) * (12 / 4);               // 5625 waves
    k_gemm_wmma<<<GRID(waves1 * 32), B, 0, stream>>>(buf1, w1hi, w1lo, buf0, NN, 256, 160, 12);

    k_node_dots<<<GRID(NN * HEADS), B, 0, stream>>>(buf0, att1, al, ar, NN, C1);

    k_fill_u32<<<GRID(NN * HC1), B, 0, stream>>>((unsigned*)buf1, 0u, NN * HC1);
    k_fill_u32<<<GRID(NN * HEADS), B, 0, stream>>>(amax, ENC_NEG_INF, NN * HEADS);
    k_fill_u32<<<GRID(NN * HEADS), B, 0, stream>>>((unsigned*)denom, 0u, NN * HEADS);

    k_edge_alpha_max<<<GRID(NT), B, 0, stream>>>(ei, NE, NT, al, ar, exbuf, amax);
    k_edge_exp<<<GRID(NT), B, 0, stream>>>(ei, NE, NT, amax, exbuf, denom);
    k_aggregate<<<GRID(NT * (HC1 / 4)), B, 0, stream>>>(ei, NE, NT, exbuf, denom,
                                                        buf0, buf1, C1, HC1);

    k_mean_bias<<<GRID(NN * C1), B, 0, stream>>>(buf1, b1, outp, NN, C1);
    #undef GRID
}
